// AdvancedInfoNCELoss_23347442221233
// MI455X (gfx1250) — compile-verified, matrix operand order checked by reading, not run on hardware
//
#include <hip/hip_runtime.h>
#include <stdint.h>

// ---------------- problem constants (from reference) ----------------
#define B_ROWS   4096
#define DIM      512
#define QSZ      32768
#define NUM_HARD 9830
#define NUM_RAND 22938
#define INV_T    (1.0f / 0.07f)

// GEMM tiling
#define BM  128
#define BN  128
#define BK  32
#define PAD 40    // padded K-chunk row (elements): 80B row stride -> conflict-free b128, 16B aligned

// ---------------- CDNA5 WMMA types ----------------
typedef __attribute__((ext_vector_type(16))) __bf16 v16bf;
typedef __attribute__((ext_vector_type(8)))  float  v8f;
typedef __attribute__((ext_vector_type(4)))  unsigned su4;
typedef __attribute__((ext_vector_type(8)))  unsigned su8;

union FragBF {
    v16bf v;
    uint4 q[2];
};
union DescG0 { unsigned w[4]; su4 v; };
union DescG1 { unsigned w[8]; su8 v; };

__device__ inline v8f wmma_bf16(v16bf a, v16bf b, v8f c) {
    // D = A(16x32 bf16) * B(32x16 bf16) + C(f32)
    return __builtin_amdgcn_wmma_f32_16x16x32_bf16(
        /*neg_a=*/false, a, /*neg_b=*/false, b,
        /*c_mod=*/(short)0, c, /*reuse_a=*/false, /*reuse_b=*/false);
}

// flat pointer -> LDS byte address (low 32 bits of the LDS aperture address)
__device__ inline unsigned ldsAddr(const void* p) {
    return (unsigned)(uintptr_t)p;
}

// ---------------------------------------------------------------------
// TDM: 2D tile load Global->LDS. One wave-level instruction moves
// tile_d1 rows x (tile_d0*4) bytes, inserting 16B of LDS padding after
// every 64B row (pad_interval=3 -> 16 DW, pad_amount=3 -> 4 DW), which
// reproduces the PAD=40 LDS layout. Tracked with TENSORcnt.
// ---------------------------------------------------------------------
__device__ inline void tdm_load_2d(unsigned lds_addr, const void* gptr,
                                   unsigned tensor_d0, unsigned tensor_d1,
                                   unsigned tile_d0, unsigned tile_d1,
                                   unsigned d0_stride)
{
    const unsigned long long ga = (unsigned long long)(uintptr_t)gptr;

    DescG0 g0;
    g0.w[0] = 1u;                                   // count=1, user descriptor
    g0.w[1] = lds_addr;                             // LDS byte address
    g0.w[2] = (unsigned)ga;                         // global_addr[31:0]
    g0.w[3] = (unsigned)(ga >> 32) | (2u << 30);    // global_addr[56:32] | type=2

    DescG1 g1;
    // data_size=2 (4B), pad_enable, pad_interval=3 (16 DW), pad_amount=3 (4 DW)
    g1.w[0] = (2u << 16) | (1u << 20) | (3u << 22) | (3u << 25);
    g1.w[1] = (tensor_d0 & 0xFFFFu) << 16;                       // tensor_dim0[15:0]
    g1.w[2] = (tensor_d0 >> 16) | ((tensor_d1 & 0xFFFFu) << 16); // dim0 hi | dim1 lo
    g1.w[3] = (tensor_d1 >> 16) | (tile_d0 << 16);               // dim1 hi | tile_dim0
    g1.w[4] = tile_d1;                                            // tile_dim1 (tile_dim2=0)
    g1.w[5] = d0_stride;                                          // tensor_dim0_stride[31:0]
    g1.w[6] = 0;                                                  // stride hi | dim1_stride lo
    g1.w[7] = 0;

    asm volatile("tensor_load_to_lds %0, %1" :: "s"(g0.v), "s"(g1.v) : "memory");
}

// monotonic float<->u32 key (unsigned order == float order)
__device__ inline unsigned fkey(float f) {
    unsigned u = __float_as_uint(f);
    return u ^ (((int)u >> 31) | 0x80000000u);
}
__device__ inline float keyToFloat(unsigned k) {
    unsigned u = (k & 0x80000000u) ? (k ^ 0x80000000u) : ~k;
    return __uint_as_float(u);
}

// =====================================================================
// Kernel 1: normalize eeg rows -> split bf16 hi/lo; pos = <eegn, clipn>
// =====================================================================
__global__ __launch_bounds__(256) void prep_eeg_clip(
    const float* __restrict__ eeg, const float* __restrict__ clip,
    __bf16* __restrict__ aHi, __bf16* __restrict__ aLo,
    float* __restrict__ pos)
{
    const int lane = threadIdx.x & 31;
    const int wave = threadIdx.x >> 5;
    const int row  = blockIdx.x * 8 + wave;

    const float* er = eeg  + (size_t)row * DIM;
    const float* cr = clip + (size_t)row * DIM;

    float e[16];
    float ss_e = 0.f, ss_c = 0.f, dt = 0.f;
#pragma unroll
    for (int j = 0; j < 16; ++j) {
        const int idx = lane + j * 32;
        const float ev = er[idx];
        const float cv = cr[idx];
        e[j] = ev;
        ss_e += ev * ev;
        ss_c += cv * cv;
        dt   += ev * cv;
    }
#pragma unroll
    for (int o = 16; o > 0; o >>= 1) {
        ss_e += __shfl_xor(ss_e, o, 32);
        ss_c += __shfl_xor(ss_c, o, 32);
        dt   += __shfl_xor(dt,   o, 32);
    }
    const float inv_e = 1.0f / fmaxf(sqrtf(ss_e), 1e-12f);
    const float inv_c = 1.0f / fmaxf(sqrtf(ss_c), 1e-12f);
    if (lane == 0) pos[row] = dt * inv_e * inv_c;

    __bf16* hr = aHi + (size_t)row * DIM;
    __bf16* lr = aLo + (size_t)row * DIM;
#pragma unroll
    for (int j = 0; j < 16; ++j) {
        const int idx = lane + j * 32;
        const float x = e[j] * inv_e;
        const __bf16 h = (__bf16)x;
        const __bf16 l = (__bf16)(x - (float)h);
        hr[idx] = h;
        lr[idx] = l;
    }
}

// =====================================================================
// Kernel 2: split (already L2-normalized) queue into bf16 hi/lo
// =====================================================================
__global__ __launch_bounds__(256) void split_queue(
    const float* __restrict__ q, __bf16* __restrict__ qHi, __bf16* __restrict__ qLo)
{
    const size_t i = ((size_t)blockIdx.x * 256 + threadIdx.x) * 4;
    const float4 v = *reinterpret_cast<const float4*>(q + i);
    const float x[4] = {v.x, v.y, v.z, v.w};
#pragma unroll
    for (int j = 0; j < 4; ++j) {
        const __bf16 h = (__bf16)x[j];
        qHi[i + j] = h;
        qLo[i + j] = (__bf16)(x[j] - (float)h);
    }
}

// =====================================================================
// Kernel 3: neg = eegn @ queue^T via split-bf16 WMMA.
// 128x128 tile / 256-thread block. K-chunks staged into LDS by the
// Tensor Data Mover (4 x tensor_load_to_lds per chunk, issued by wave 0,
// TENSORcnt double-buffered); WMMA fed by ds_load_b128.
// =====================================================================
__global__ __launch_bounds__(256) void gemm_neg(
    const __bf16* __restrict__ aHi, const __bf16* __restrict__ aLo,
    const __bf16* __restrict__ qHi, const __bf16* __restrict__ qLo,
    float* __restrict__ neg)
{
    __shared__ __bf16 sAH[2][BM][PAD];
    __shared__ __bf16 sAL[2][BM][PAD];
    __shared__ __bf16 sBH[2][BN][PAD];
    __shared__ __bf16 sBL[2][BN][PAD];

    const int tid  = threadIdx.x;
    const int lane = tid & 31;
    const int wave = tid >> 5;
    const int m    = lane & 15;   // M within A tile / N within B tile
    const int h    = lane >> 4;   // K-half selector (documented 16-bit layout)

    const int rowbase = blockIdx.y * BM;
    const int colbase = blockIdx.x * BN;

    const __bf16* gAH = aHi + (size_t)rowbase * DIM;
    const __bf16* gAL = aLo + (size_t)rowbase * DIM;
    const __bf16* gBH = qHi + (size_t)colbase * DIM;
    const __bf16* gBL = qLo + (size_t)colbase * DIM;

    // one K-chunk = 128 rows x 64B, 4 matrices -> 4 TDM descriptors
    auto tdm_stage = [&](int buf, int kk) {
        const unsigned d0   = DIM / 2;   // 256: row length in 4B units
        const unsigned str  = DIM / 2;   // row stride in 4B units
        const unsigned td0  = BK / 2;    // 16:  tile row (64B) in 4B units
        tdm_load_2d(ldsAddr(&sAH[buf][0][0]), (const char*)gAH + kk * 2, d0, BM, td0, BM, str);
        tdm_load_2d(ldsAddr(&sAL[buf][0][0]), (const char*)gAL + kk * 2, d0, BM, td0, BM, str);
        tdm_load_2d(ldsAddr(&sBH[buf][0][0]), (const char*)gBH + kk * 2, d0, BN, td0, BN, str);
        tdm_load_2d(ldsAddr(&sBL[buf][0][0]), (const char*)gBL + kk * 2, d0, BN, td0, BN, str);
    };

    v8f acc[8] = {};

    if (wave == 0) tdm_stage(0, 0);

    const int NC = DIM / BK; // 16
    for (int c = 0; c < NC; ++c) {
        const int buf = c & 1;
        if (wave == 0) {
            if (c + 1 < NC) {
                tdm_stage(buf ^ 1, (c + 1) * BK);
                // 4 newest TDM ops belong to next buffer; <=4 => this buffer done
                __builtin_amdgcn_s_wait_tensorcnt(4);
            } else {
                __builtin_amdgcn_s_wait_tensorcnt(0);
            }
        }
        __syncthreads();

        FragBF ah, al;
        {
            const int r = wave * 16 + m;
            ah.q[0] = *reinterpret_cast<const uint4*>(&sAH[buf][r][h * 8]);
            ah.q[1] = *reinterpret_cast<const uint4*>(&sAH[buf][r][16 + h * 8]);
            al.q[0] = *reinterpret_cast<const uint4*>(&sAL[buf][r][h * 8]);
            al.q[1] = *reinterpret_cast<const uint4*>(&sAL[buf][r][16 + h * 8]);
        }
#pragma unroll
        for (int t = 0; t < 8; ++t) {
            const int r = t * 16 + m;
            FragBF bh, bl;
            bh.q[0] = *reinterpret_cast<const uint4*>(&sBH[buf][r][h * 8]);
            bh.q[1] = *reinterpret_cast<const uint4*>(&sBH[buf][r][16 + h * 8]);
            bl.q[0] = *reinterpret_cast<const uint4*>(&sBL[buf][r][h * 8]);
            bl.q[1] = *reinterpret_cast<const uint4*>(&sBL[buf][r][16 + h * 8]);
            acc[t] = wmma_bf16(ah.v, bh.v, acc[t]);   // hi*hi
            acc[t] = wmma_bf16(ah.v, bl.v, acc[t]);   // hi*lo
            acc[t] = wmma_bf16(al.v, bh.v, acc[t]);   // lo*hi
        }
        __syncthreads();   // all waves done reading before buffer reuse
    }

    // C/D layout: VGPR r -> M = r + 8*h, N = lane&15
#pragma unroll
    for (int t = 0; t < 8; ++t) {
#pragma unroll
        for (int r = 0; r < 8; ++r) {
            const int mo = rowbase + wave * 16 + h * 8 + r;
            const int no = colbase + t * 16 + m;
            neg[(size_t)mo * QSZ + no] = acc[t][r];
        }
    }
}

// =====================================================================
// Kernel 4: per-row top-K selection + logsumexp + accuracy.
// Row (128KB) staged into LDS via global_load_async_to_lds_b128.
// =====================================================================
__global__ __launch_bounds__(1024) void row_select(
    const float* __restrict__ neg, const float* __restrict__ pos,
    const int* __restrict__ rind,
    float* __restrict__ lossRows, float* __restrict__ accRows)
{
    extern __shared__ unsigned char smem[];
    float*    vals = reinterpret_cast<float*>(smem);                         // QSZ
    unsigned* hist = reinterpret_cast<unsigned*>(smem + QSZ * 4);            // 256
    float*    red  = reinterpret_cast<float*>(smem + QSZ * 4 + 256 * 4);     // 32
    unsigned* sh   = reinterpret_cast<unsigned*>(smem + QSZ * 4 + 256 * 4 + 32 * 4); // 2

    const int tid = threadIdx.x;
    const int row = blockIdx.x;
    const float* rowp = neg + (size_t)row * QSZ;
    const unsigned vbase = ldsAddr(&vals[0]);

    // ---- async bulk copy: 8 x (1024 lanes x 16B) = 128KB -> LDS ----
#pragma unroll
    for (int it = 0; it < 8; ++it) {
        const unsigned byteOff = (unsigned)(tid + it * 1024) * 16u;
        const unsigned l = vbase + byteOff;
        asm volatile("global_load_async_to_lds_b128 %0, %1, %2"
                     :: "v"(l), "v"(byteOff), "s"(rowp) : "memory");
    }
    asm volatile("s_wait_asynccnt 0x0" ::: "memory");
    __syncthreads();

    // ---- row max ----
    float lmax = -3.4e38f;
    for (int i = tid; i < QSZ; i += 1024) lmax = fmaxf(lmax, vals[i]);
#pragma unroll
    for (int o = 16; o > 0; o >>= 1) lmax = fmaxf(lmax, __shfl_xor(lmax, o, 32));
    if ((tid & 31) == 0) red[tid >> 5] = lmax;
    __syncthreads();
    if (tid < 32) {
        float x = red[tid];
#pragma unroll
        for (int o = 16; o > 0; o >>= 1) x = fmaxf(x, __shfl_xor(x, o, 32));
        if (tid == 0) red[0] = x;
    }
    __syncthreads();
    const float rowmax = red[0];
    const float p      = pos[row];
    const float ML     = fmaxf(p, rowmax) * INV_T;   // max logit, stabilization
    __syncthreads();

    // ---- radix select: K-th largest over monotone keys ----
    unsigned prefix = 0, remaining = NUM_HARD;
    for (int shift = 24; shift >= 0; shift -= 8) {
        if (tid < 256) hist[tid] = 0;
        __syncthreads();
        const unsigned pmask = (shift == 24) ? 0u : (0xFFFFFFFFu << (shift + 8));
        for (int i = tid; i < QSZ; i += 1024) {
            const unsigned k = fkey(vals[i]);
            if ((k & pmask) == prefix)
                atomicAdd(&hist[(k >> shift) & 255u], 1u);
        }
        __syncthreads();
        if (tid == 0) {
            unsigned cum = 0;
            int b = 255;
            for (; b > 0; --b) {
                const unsigned c = hist[b];
                if (cum + c >= remaining) break;
                cum += c;
            }
            sh[0] = prefix | ((unsigned)b << shift);
            sh[1] = remaining - cum;
        }
        __syncthreads();
        prefix = sh[0];
        remaining = sh[1];   // #copies of threshold value inside top-K
        __syncthreads();
    }
    const unsigned thr   = prefix;
    const float   thrval = keyToFloat(thr);

    // ---- exp-sums: strict top (> thr) + random gather from LDS ----
    float accum = 0.f;
    for (int i = tid; i < QSZ; i += 1024) {
        const float v = vals[i];
        if (fkey(v) > thr) accum += expf(v * INV_T - ML);
    }
    const int* rp = rind + (size_t)row * NUM_RAND;
    for (int j = tid; j < NUM_RAND; j += 1024) {
        const int idx = rp[j];
        accum += expf(vals[idx] * INV_T - ML);
    }
#pragma unroll
    for (int o = 16; o > 0; o >>= 1) accum += __shfl_xor(accum, o, 32);
    __syncthreads();
    if ((tid & 31) == 0) red[tid >> 5] = accum;
    __syncthreads();
    if (tid == 0) {
        float total = 0.f;
        for (int w = 0; w < 32; ++w) total += red[w];
        total += (float)remaining * expf(thrval * INV_T - ML); // ties at threshold
        total += expf(p * INV_T - ML);                          // positive logit
        const float logz = ML + logf(total);
        lossRows[row] = logz - p * INV_T;
        accRows[row]  = (p >= rowmax) ? 1.0f : 0.0f;            // argmax == 0
    }
}

// =====================================================================
// Kernel 5: mean over rows -> out[0]=loss, out[1]=accuracy
// =====================================================================
__global__ __launch_bounds__(256) void finalize(
    const float* __restrict__ lossRows, const float* __restrict__ accRows,
    float* __restrict__ out)
{
    __shared__ float s1[256], s2[256];
    const int tid = threadIdx.x;
    float a = 0.f, b = 0.f;
    for (int i = tid; i < B_ROWS; i += 256) { a += lossRows[i]; b += accRows[i]; }
    s1[tid] = a; s2[tid] = b;
    __syncthreads();
    for (int o = 128; o > 0; o >>= 1) {
        if (tid < o) { s1[tid] += s1[tid + o]; s2[tid] += s2[tid + o]; }
        __syncthreads();
    }
    if (tid == 0) {
        out[0] = s1[0] / (float)B_ROWS;
        out[1] = s2[0] / (float)B_ROWS;
    }
}

// =====================================================================
// host launcher
// =====================================================================
extern "C" void kernel_launch(void* const* d_in, const int* in_sizes, int n_in,
                              void* d_out, int out_size, void* d_ws, size_t ws_size,
                              hipStream_t stream)
{
    (void)in_sizes; (void)n_in; (void)out_size; (void)ws_size;

    const float* eeg   = (const float*)d_in[0];
    const float* clip  = (const float*)d_in[1];
    const float* queue = (const float*)d_in[2];
    const int*   rind  = (const int*)d_in[3];
    float*       out   = (float*)d_out;

    char* ws = (char*)d_ws;
    size_t off = 0;
    auto alloc = [&](size_t bytes) -> void* {
        void* ptr = (void*)(ws + off);
        off += (bytes + 255) & ~(size_t)255;
        return ptr;
    };
    // total ~584 MB of workspace (neg matrix dominates at 512 MB)
    __bf16* aHi      = (__bf16*)alloc((size_t)B_ROWS * DIM * sizeof(__bf16));
    __bf16* aLo      = (__bf16*)alloc((size_t)B_ROWS * DIM * sizeof(__bf16));
    __bf16* qHi      = (__bf16*)alloc((size_t)QSZ * DIM * sizeof(__bf16));
    __bf16* qLo      = (__bf16*)alloc((size_t)QSZ * DIM * sizeof(__bf16));
    float*  pos      = (float*)alloc((size_t)B_ROWS * sizeof(float));
    float*  lossRows = (float*)alloc((size_t)B_ROWS * sizeof(float));
    float*  accRows  = (float*)alloc((size_t)B_ROWS * sizeof(float));
    float*  neg      = (float*)alloc((size_t)B_ROWS * QSZ * sizeof(float));

    prep_eeg_clip<<<B_ROWS / 8, 256, 0, stream>>>(eeg, clip, aHi, aLo, pos);
    split_queue<<<(QSZ * DIM) / 1024, 256, 0, stream>>>(queue, qHi, qLo);
    gemm_neg<<<dim3(QSZ / BN, B_ROWS / BM), 256, 0, stream>>>(aHi, aLo, qHi, qLo, neg);

    const size_t smem = (size_t)QSZ * 4 + 256 * 4 + 32 * 4 + 2 * 4; // ~132 KB < 320 KB WGP LDS
    row_select<<<B_ROWS, 1024, smem, stream>>>(neg, pos, rind, lossRows, accRows);

    finalize<<<1, 256, 0, stream>>>(lossRows, accRows, out);
}
